// Attention_16896401343184
// MI455X (gfx1250) — matrix operand order repaired and checked by reading, not compiled, not run on hardware
//
#include <hip/hip_runtime.h>
#include <hip/hip_bf16.h>

// ---------------------------------------------------------------------------
// MI455X (gfx1250) fused attention layer.
// ~69 GFLOP vs ~50MB HBM traffic -> compute bound; all matmuls on
// v_wmma_f32_16x16x32_f16, flash-attention tiling, TDM (tensor_load_to_lds)
// for the contiguous K tile when the toolchain exposes the builtin.
// Workspace layout (bytes), needs 48 MiB:
//   [ 0M, 8M)  xn_h   f16 [4096,1024]   RMSNorm(x)
//   [ 8M,16M)  w_h    f16 wq|wk|wv|wo (1M elems each)
//   [16M,24M)  q_h    f16 [B,H,S,Dh]  (post-RoPE)
//   [24M,32M)  k_h    f16 [B,H,S,Dh]  (post-RoPE)
//   [32M,40M)  v_h    f16 [B,H,S,Dh]
//   [40M,48M)  attn_h f16 [4096,1024]
// ---------------------------------------------------------------------------

typedef _Float16 v8h  __attribute__((ext_vector_type(8)));
typedef _Float16 v16h __attribute__((ext_vector_type(16)));
typedef float    v8f  __attribute__((ext_vector_type(8)));
typedef unsigned int v4u __attribute__((ext_vector_type(4)));
typedef int          v4i __attribute__((ext_vector_type(4)));
typedef int          v8i __attribute__((ext_vector_type(8)));

#if defined(__gfx1250__) && __has_builtin(__builtin_amdgcn_tensor_load_to_lds) && \
    __has_builtin(__builtin_amdgcn_s_wait_tensorcnt)
#define HAVE_TDM 1
#else
#define HAVE_TDM 0
#endif
#if __has_include(<hip/amd_detail/amd_gfx1250_TDM.h>)
#define TDM_6ARG 1
#else
#define TDM_6ARG 0
#endif

__device__ __forceinline__ v16h cat16(v8h lo, v8h hi) {
  return __builtin_shufflevector(lo, hi, 0,1,2,3,4,5,6,7,8,9,10,11,12,13,14,15);
}
// A-fragment (16x32 f16): lane m=lane&15; halves [0..15]=K(hf*16+0..15),
// hf=lane>>4.  p points at row m, K-chunk base.
__device__ __forceinline__ v16h ldA(const _Float16* p, int hf) {
  return cat16(*(const v8h*)(p + hf*16), *(const v8h*)(p + hf*16 + 8));
}
// B-fragment (32x16 f16): lane col n=lane&15 holds 16 contiguous K at
// kbase=(lane>>4)*16; p points at the 16-half run.
__device__ __forceinline__ v16h ldB(const _Float16* p) {
  return cat16(*(const v8h*)p, *(const v8h*)(p + 8));
}
__device__ __forceinline__ v8f wmma_f16(v16h a, v16h b, v8f c) {
  return __builtin_amdgcn_wmma_f32_16x16x32_f16(false, a, false, b,
                                                (short)0, c, false, false);
}

// ------------------------- weight fp32 -> f16 ------------------------------
__global__ __launch_bounds__(256) void cvtw_k(
    const float* __restrict__ w0, const float* __restrict__ w1,
    const float* __restrict__ w2, const float* __restrict__ w3,
    _Float16* __restrict__ dst)
{
  const float* srcs[4] = {w0, w1, w2, w3};
  const float* s = srcs[blockIdx.y];
  _Float16* d = dst + (size_t)blockIdx.y * (1u << 20);
  int i = (blockIdx.x * 256 + threadIdx.x) * 4;
  float4 v = *(const float4*)(s + i);
  d[i+0] = (_Float16)v.x; d[i+1] = (_Float16)v.y;
  d[i+2] = (_Float16)v.z; d[i+3] = (_Float16)v.w;
}

// ------------------------- RMSNorm + cvt -----------------------------------
__global__ __launch_bounds__(256) void rmsnorm_k(
    const float* __restrict__ x, const float* __restrict__ w,
    _Float16* __restrict__ xh)
{
  const int row = blockIdx.x, tid = threadIdx.x;
  const float* xr = x + (size_t)row * 1024;
  float4 v = ((const float4*)xr)[tid];
  float ss = v.x*v.x + v.y*v.y + v.z*v.z + v.w*v.w;
  #pragma unroll
  for (int m = 16; m >= 1; m >>= 1) ss += __shfl_xor(ss, m, 32);
  __shared__ float red[8];
  if ((tid & 31) == 0) red[tid >> 5] = ss;
  __syncthreads();
  float tot = 0.f;
  #pragma unroll
  for (int i = 0; i < 8; ++i) tot += red[i];
  float rinv = rsqrtf(tot * (1.0f / 1024.0f) + 1.1920929e-07f);
  float4 wv = ((const float4*)w)[tid];
  _Float16* o = xh + (size_t)row * 1024 + tid * 4;
  o[0] = (_Float16)(v.x * rinv * wv.x); o[1] = (_Float16)(v.y * rinv * wv.y);
  o[2] = (_Float16)(v.z * rinv * wv.z); o[3] = (_Float16)(v.w * rinv * wv.w);
}

// ------------------------- 128x64-tile GEMM (double-buffered) --------------
// C[4096,1024] = A[4096,1024] @ W[1024,1024], f16 in, f32 acc.
// EPI: 0 = RoPE + store f16 [B,H,S,Dh];  2 = store f16 [B,H,S,Dh];
//      3 = store f32 [4096,1024].
template<int EPI>
__global__ __launch_bounds__(256) void gemm_k(
    const _Float16* __restrict__ A, const _Float16* __restrict__ W,
    _Float16* __restrict__ outH, float* __restrict__ outF)
{
  constexpr int K = 1024, N = 1024, S = 2048;
  __shared__ _Float16 lA [2][128 * 32];   // A tile, row-major [m][k]
  __shared__ _Float16 lWt[2][64 * 32];    // W tile transposed [n][k]
  const int tid = threadIdx.x, lane = tid & 31, wid = tid >> 5;
  const int hf = lane >> 4, l15 = lane & 15;
  const int m0 = blockIdx.x * 128, n0 = blockIdx.y * 64;
  const int wm = (wid >> 1) * 32, wn = (wid & 1) * 32;

  v8f zero = {0.f,0.f,0.f,0.f,0.f,0.f,0.f,0.f};
  v8f acc[2][2] = {{zero, zero}, {zero, zero}};

  const int arow = tid >> 1, akh = (tid & 1) * 16;   // A: 16 halves/thread
  const int wkr = tid >> 3, wn8 = (tid & 7) * 8;     // W: 8 halves/thread

  // preload tile kb=0 into buffer 0
  {
    const _Float16* as = A + (size_t)(m0 + arow) * K + akh;
    *(v8h*)&lA[0][arow*32 + akh]     = *(const v8h*)as;
    *(v8h*)&lA[0][arow*32 + akh + 8] = *(const v8h*)(as + 8);
    v8h wv = *(const v8h*)(W + (size_t)wkr * N + n0 + wn8);
    #pragma unroll
    for (int e = 0; e < 8; ++e) lWt[0][(wn8 + e)*32 + wkr] = wv[e];
  }

  int buf = 0;
  for (int kb = 0; kb < K; kb += 32) {
    __syncthreads();
    const bool nxt = (kb + 32) < K;
    v8h a0, a1, wv;
    if (nxt) {  // global loads for next tile fly while WMMAs run
      const _Float16* as = A + (size_t)(m0 + arow) * K + kb + 32 + akh;
      a0 = *(const v8h*)as;
      a1 = *(const v8h*)(as + 8);
      wv = *(const v8h*)(W + (size_t)(kb + 32 + wkr) * N + n0 + wn8);
    }
    v16h af[2], bf[2];
    #pragma unroll
    for (int i = 0; i < 2; ++i) af[i] = ldA(&lA[buf][(wm + i*16 + l15)*32], hf);
    #pragma unroll
    for (int i = 0; i < 2; ++i) bf[i] = ldB(&lWt[buf][(wn + i*16 + l15)*32 + hf*16]);
    #pragma unroll
    for (int mm = 0; mm < 2; ++mm)
      #pragma unroll
      for (int nn = 0; nn < 2; ++nn)
        acc[mm][nn] = wmma_f16(af[mm], bf[nn], acc[mm][nn]);
    if (nxt) {
      *(v8h*)&lA[buf^1][arow*32 + akh]     = a0;
      *(v8h*)&lA[buf^1][arow*32 + akh + 8] = a1;
      #pragma unroll
      for (int e = 0; e < 8; ++e) lWt[buf^1][(wn8 + e)*32 + wkr] = wv[e];
    }
    buf ^= 1;
  }

  // Epilogue.  C layout: lanes 0-15 hold (M=r, N=lane); 16-31 (M=r+8, N=lane-16).
  #pragma unroll
  for (int mm = 0; mm < 2; ++mm) {
    #pragma unroll
    for (int nn = 0; nn < 2; ++nn) {
      #pragma unroll
      for (int r = 0; r < 8; ++r) {
        float val = acc[mm][nn][r];
        int gm = m0 + wm + mm*16 + r + 8*hf;
        int gn = n0 + wn + nn*16 + l15;
        if constexpr (EPI == 3) {
          outF[(size_t)gm * N + gn] = val;
        } else if constexpr (EPI == 2) {
          int s = gm & (S - 1), bb = gm >> 11;
          int hh = gn >> 6,     d  = gn & 63;
          outH[(((size_t)(bb*16 + hh)) * S + s) * 64 + d] = (_Float16)val;
        } else {  // RoPE: partner element lives in lane^1 (N differs by 1)
          float partner = __shfl_xor(val, 1, 32);
          int s = gm & (S - 1), bb = gm >> 11;
          int hh = gn >> 6,     d  = gn & 63;
          int p = d >> 1;
          float freq = __expf(-9.2103403719761836f * ((float)(2*p) * (1.0f/64.0f)));
          float ang = (float)s * freq;
          float cs = __cosf(ang), sn = __sinf(ang);
          float rot = (d & 1) ? (val*cs + partner*sn) : (val*cs - partner*sn);
          outH[(((size_t)(bb*16 + hh)) * S + s) * 64 + d] = (_Float16)rot;
        }
      }
    }
  }
}

// ------------------------- flash attention ---------------------------------
// grid (S/128, H, B), 256 threads. Each wave owns 16 Q rows; block stages
// 64-key K (via TDM if available) and V^T tiles in LDS; online softmax;
// P@V accumulated via WMMA.
__global__ __launch_bounds__(256) void flash_k(
    const _Float16* __restrict__ Qh, const _Float16* __restrict__ Kh,
    const _Float16* __restrict__ Vh, _Float16* __restrict__ attnH)
{
  constexpr int S = 2048, Dh = 64, H = 16;
  __shared__ _Float16 lK [64 * 64];   // [key][d]
  __shared__ _Float16 lVt[64 * 64];   // [d][key]
  __shared__ _Float16 lP [8 * 16 * 64];  // per-wave P scratch [m][key]
  const int tid = threadIdx.x, lane = tid & 31, wid = tid >> 5;
  const int hf = lane >> 4, l15 = lane & 15;
  const int q0 = blockIdx.x * 128;
  const int h = blockIdx.y, b = blockIdx.z;
  const size_t bh = ((size_t)(b*H + h)) * S * Dh;
  const _Float16* Qb = Qh + bh;
  const _Float16* Kb = Kh + bh;
  const _Float16* Vb = Vh + bh;
  const int qrow0 = q0 + wid * 16;
  _Float16* lPw = lP + wid * (16 * 64);

  v16h aq[2];
  #pragma unroll
  for (int kc = 0; kc < 2; ++kc)
    aq[kc] = ldA(Qb + (size_t)(qrow0 + l15) * Dh + kc*32, hf);

  v8f zero = {0.f,0.f,0.f,0.f,0.f,0.f,0.f,0.f};
  v8f O[4] = {zero, zero, zero, zero};
  float mi[8], li[8];
  #pragma unroll
  for (int r = 0; r < 8; ++r) { mi[r] = -__builtin_inff(); li[r] = 0.f; }

  const int nkb = q0 / 64 + 2;               // causal: keys <= q-tile end
  for (int j = 0; j < nkb; ++j) {
    const int j0 = j * 64;
    __syncthreads();
#if HAVE_TDM
    // --- K tile via Tensor Data Mover: 64x64 f16 tile is one contiguous ---
    // --- 8KB run -> 1D descriptor (dim0 = tile0 = 4096 elems, 2B each). ---
    if (wid == 0) {
      unsigned lkoff = (unsigned)(unsigned long long)(const void*)lK;     // LDS aperture: low 32b = LDS addr
      unsigned long long ga = (unsigned long long)(const void*)(Kb + (size_t)j0 * Dh);
      v4u g0 = { 1u,                                    // count=1, user mode
                 lkoff,                                 // D#.lds_addr
                 (unsigned)(ga & 0xffffffffu),          // global_addr[31:0]
                 (unsigned)((ga >> 32) & 0x01ffffffu) | (2u << 30) };  // [56:32] | type=2
      v8i g1 = { (int)(1u << 16),          // workgroup_mask=0, data_size=1 (2B)
                 (int)(4096u << 16),       // tensor_dim0[15:0]=4096 (bits 63:48)
                 (int)(1u << 16),          // tensor_dim0[31:16]=0, tensor_dim1[15:0]=1
                 (int)(4096u << 16),       // tensor_dim1[31:16]=0, tile_dim0=4096
                 1,                        // tile_dim1=1, tile_dim2=0
                 4096,                     // tensor_dim0_stride[31:0]
                 0, 0 };                   // strides hi = 0
      v4i gz = {0, 0, 0, 0};
#if TDM_6ARG
      v8i gz8 = {0,0,0,0,0,0,0,0};
      __builtin_amdgcn_tensor_load_to_lds(g0, g1, gz, gz, gz8, 0);
#else
      __builtin_amdgcn_tensor_load_to_lds(g0, g1, gz, gz, 0);
#endif
      __builtin_amdgcn_s_wait_tensorcnt(0);
    }
#endif
    {  // stage V transposed into LDS (and K manually if no TDM)
      int r = tid >> 2, o = (tid & 3) * 16;
#if !HAVE_TDM
      const _Float16* ks = Kb + (size_t)(j0 + r) * Dh + o;
      *(v8h*)&lK[r*64 + o]     = *(const v8h*)ks;
      *(v8h*)&lK[r*64 + o + 8] = *(const v8h*)(ks + 8);
#endif
      const _Float16* vs = Vb + (size_t)(j0 + r) * Dh + o;
      v8h v0 = *(const v8h*)vs, v1 = *(const v8h*)(vs + 8);
      #pragma unroll
      for (int e = 0; e < 8; ++e) {
        lVt[(o + e)*64 + r]     = v0[e];
        lVt[(o + 8 + e)*64 + r] = v1[e];
      }
    }
    __syncthreads();

    // S = Q @ K^T   (4 score tiles of 16x16, K-dim 64 in two 32-chunks)
    v8f sc[4];
    #pragma unroll
    for (int t = 0; t < 4; ++t) {
      v8f s = zero;
      s = wmma_f16(aq[0], ldB(&lK[(t*16 + l15)*64 +      hf*16]), s);
      s = wmma_f16(aq[1], ldB(&lK[(t*16 + l15)*64 + 32 + hf*16]), s);
      sc[t] = s;
    }
    // scale + causal mask
    #pragma unroll
    for (int t = 0; t < 4; ++t)
      #pragma unroll
      for (int r = 0; r < 8; ++r) {
        int qi = qrow0 + r + 8*hf;
        int kj = j0 + t*16 + l15;
        sc[t][r] = (kj <= qi) ? sc[t][r] * 0.125f : -3.0e38f;
      }
    // online softmax (row values live across the 16 lanes of each half-wave)
    #pragma unroll
    for (int r = 0; r < 8; ++r) {
      float rm = fmaxf(fmaxf(sc[0][r], sc[1][r]), fmaxf(sc[2][r], sc[3][r]));
      #pragma unroll
      for (int m = 8; m >= 1; m >>= 1) rm = fmaxf(rm, __shfl_xor(rm, m, 32));
      float mnew = fmaxf(mi[r], rm);
      float alpha = __expf(mi[r] - mnew);
      float rs = 0.f;
      #pragma unroll
      for (int t = 0; t < 4; ++t) { sc[t][r] = __expf(sc[t][r] - mnew); rs += sc[t][r]; }
      #pragma unroll
      for (int m = 8; m >= 1; m >>= 1) rs += __shfl_xor(rs, m, 32);
      li[r] = li[r] * alpha + rs;
      mi[r] = mnew;
      #pragma unroll
      for (int t = 0; t < 4; ++t) O[t][r] *= alpha;
    }
    // P (C-layout) -> per-wave LDS -> A-fragments, then O += P @ V
    #pragma unroll
    for (int t = 0; t < 4; ++t)
      #pragma unroll
      for (int r = 0; r < 8; ++r)
        lPw[(r + 8*hf)*64 + t*16 + l15] = (_Float16)sc[t][r];
    v16h aP0 = ldA(lPw + l15*64,      hf);
    v16h aP1 = ldA(lPw + l15*64 + 32, hf);
    #pragma unroll
    for (int t = 0; t < 4; ++t) {
      O[t] = wmma_f16(aP0, ldB(&lVt[(t*16 + l15)*64 +      hf*16]), O[t]);
      O[t] = wmma_f16(aP1, ldB(&lVt[(t*16 + l15)*64 + 32 + hf*16]), O[t]);
    }
  }

  // epilogue: divide by l_i, store f16 [B*S, 1024] for the output GEMM
  #pragma unroll
  for (int r = 0; r < 8; ++r) {
    float inv = 1.0f / li[r];
    int srow = qrow0 + r + 8*hf;
    size_t grow = (size_t)b * S + srow;
    #pragma unroll
    for (int t = 0; t < 4; ++t) {
      int col = h * Dh + t*16 + l15;
      attnH[grow * 1024 + col] = (_Float16)(O[t][r] * inv);
    }
  }
}

// ---------------------------------------------------------------------------
extern "C" void kernel_launch(void* const* d_in, const int* in_sizes, int n_in,
                              void* d_out, int out_size, void* d_ws, size_t ws_size,
                              hipStream_t stream) {
  const float* x  = (const float*)d_in[0];
  const float* nw = (const float*)d_in[1];
  const float* wq = (const float*)d_in[2];
  const float* wk = (const float*)d_in[3];
  const float* wv = (const float*)d_in[4];
  const float* wo = (const float*)d_in[5];
  float* out = (float*)d_out;

  char* ws = (char*)d_ws;                       // needs 48 MiB
  _Float16* xh  = (_Float16*)(ws);
  _Float16* wqh = (_Float16*)(ws + (8ull  << 20));
  _Float16* wkh = wqh + (1u << 20);
  _Float16* wvh = wqh + (2u << 20);
  _Float16* woh = wqh + (3u << 20);
  _Float16* qh  = (_Float16*)(ws + (16ull << 20));
  _Float16* kh  = (_Float16*)(ws + (24ull << 20));
  _Float16* vh  = (_Float16*)(ws + (32ull << 20));
  _Float16* ah  = (_Float16*)(ws + (40ull << 20));

  cvtw_k   <<<dim3(1024, 4), 256, 0, stream>>>(wq, wk, wv, wo, wqh);
  rmsnorm_k<<<4096,          256, 0, stream>>>(x, nw, xh);
  gemm_k<0><<<dim3(32, 16),  256, 0, stream>>>(xh, wqh, qh, nullptr);  // Q + RoPE
  gemm_k<0><<<dim3(32, 16),  256, 0, stream>>>(xh, wkh, kh, nullptr);  // K + RoPE
  gemm_k<2><<<dim3(32, 16),  256, 0, stream>>>(xh, wvh, vh, nullptr);  // V
  flash_k  <<<dim3(16, 16, 2), 256, 0, stream>>>(qh, kh, vh, ah);
  gemm_k<3><<<dim3(32, 16),  256, 0, stream>>>(ah, woh, nullptr, out); // @ wo
}
